// PatternMatrix_80616536145987
// MI455X (gfx1250) — matrix operand order repaired and checked by reading, not compiled, not run on hardware
//
#include <hip/hip_runtime.h>
#include <hip/hip_bf16.h>
#include <stdint.h>

// ---------------------------------------------------------------------------
// PatternMatrix.smooth_sequence for MI455X (gfx1250, wave32).
//
//   score_k(t) = logM[prev,k] + logMT[next,k] - log S[prev,next] + gumbel(t,k)
//   sample(t)  = argmax_k score_k(t)           (Gumbel-max categorical)
//
// S = M @ M via v_wmma_f32_16x16x32_bf16. GEMM A-bands are staged into LDS
// once per workgroup with CDNA5 async load-to-LDS (ASYNCcnt), cutting A-side
// L2 traffic 8x (8 waves share one 16x512 band).
// ---------------------------------------------------------------------------

#define KSYM 512

typedef __attribute__((ext_vector_type(16))) __bf16 v16bf;
typedef __attribute__((ext_vector_type(8)))  float  v8f;

// ---------------- pass 1: bf16 copies + transposes + log tables -------------
__global__ void pm_prep(const float* __restrict__ M,
                        __bf16* __restrict__ Mbf, __bf16* __restrict__ MTbf,
                        float* __restrict__ logM, float* __restrict__ logMT) {
    int idx = blockIdx.x * blockDim.x + threadIdx.x;   // 0 .. K*K-1
    int i = idx >> 9;
    int j = idx & (KSYM - 1);
    float m = M[idx];
    __bf16 h = (__bf16)m;
    Mbf[idx]           = h;
    MTbf[j * KSYM + i] = h;
    float lm = __logf(fmaxf(m, 1e-30f));
    logM[idx]           = lm;
    logMT[j * KSYM + i] = lm;
}

// ---------------- pass 2: S = M @ M via WMMA bf16 ---------------------------
// One workgroup (8 waves) handles tile-row tr and 8 consecutive 16x16 output
// tiles. The shared 16x512 bf16 A band (16 KB) is staged into LDS with async
// load-to-LDS; each wave then runs the K-loop with A fragments from LDS and
// B fragments (its private columns) from global MTbf.
// Fragment layouts per ISA 7.12.2 (16-bit A 16x32, B 32x16, f32 C/D 16x16).
__global__ void pm_gemm_bf16(const __bf16* __restrict__ A,
                             const __bf16* __restrict__ BT,
                             float* __restrict__ S) {
    __shared__ __bf16 sA[16 * KSYM];                  // 16 KB band

    const int tid  = threadIdx.x;
    const int lane = tid & 31;
    const int wid  = tid >> 5;                        // wave in block, 0..7
    const int tr   = blockIdx.x >> 2;                 // tile row 0..31
    const int tc   = (blockIdx.x & 3) * 8 + wid;      // tile col 0..31

    // ---- async-stage A band: 16 rows x 512 cols bf16 = 1024 x 16B chunks ---
    {
        const __bf16* gbase = A + (size_t)tr * 16 * KSYM;
#pragma unroll
        for (int i = 0; i < 4; ++i) {
            int c = i * 256 + tid;                    // 16B chunk index
            unsigned lds_off = (unsigned)(uintptr_t)(&sA[c * 8]);
            const __bf16* gsrc = gbase + c * 8;
            asm volatile("global_load_async_to_lds_b128 %0, %1, off"
                         :: "v"(lds_off), "v"(gsrc) : "memory");
        }
        asm volatile("s_wait_asynccnt 0x0" ::: "memory");
    }
    __syncthreads();

    const int lh  = (lane >> 4) & 1;                  // lane half
    const int l15 = lane & 15;
    const int bcol = tc * 16 + l15;                   // B column for this lane

    v8f c = {};
    for (int kk = 0; kk < KSYM; kk += 32) {
        v16bf a, b;
        // A fragment from LDS (row = l15 within band)
        const __bf16* ap = &sA[l15 * KSYM + kk + lh * 8];
#pragma unroll
        for (int i = 0; i < 8; ++i) { a[i] = ap[i]; a[8 + i] = ap[16 + i]; }
        // B fragment from global (column-contiguous via MTbf)
        const __bf16* bp = BT + bcol * KSYM + kk + lh * 8;
#pragma unroll
        for (int i = 0; i < 8; ++i) { b[i] = bp[i]; b[8 + i] = bp[16 + i]; }
        c = __builtin_amdgcn_wmma_f32_16x16x32_bf16(
                /*neg_a=*/false, a, /*neg_b=*/false, b,
                /*c_mod=*/(short)0, c, /*reuse_a=*/false, /*reuse_b=*/false);
    }

    // C/D layout: VGPR r -> (M = lh*8 + r, N = lane&15)
    const int n = bcol;
    const int mbase = tr * 16 + lh * 8;
#pragma unroll
    for (int r = 0; r < 8; ++r) S[(mbase + r) * KSYM + n] = c[r];
}

// ---------------- pass 3: Gumbel-max sampling, one wave32 per position ------
__device__ __forceinline__ float pm_u01(uint32_t x) {
    // SplitMix32 finalizer -> uniform in (0,1)
    x ^= x >> 16; x *= 0x7feb352dU;
    x ^= x >> 15; x *= 0x846ca68bU;
    x ^= x >> 16;
    return (float)(x >> 8) * (1.0f / 16777216.0f) + (0.5f / 16777216.0f);
}

__global__ void pm_sample(const int* __restrict__ z,
                          const unsigned char* __restrict__ mask,
                          const float* __restrict__ marg,
                          const float* __restrict__ logM,
                          const float* __restrict__ logMT,
                          const float* __restrict__ S,
                          int* __restrict__ out, int T) {
    int wave = (blockIdx.x * blockDim.x + threadIdx.x) >> 5;
    int lane = threadIdx.x & 31;
    int t = wave;
    if (t >= T) return;

    int zt = z[t];
    if (mask[t] == 0) {                 // unmasked: pass through
        if (lane == 0) out[t] = zt;
        return;
    }
    bool prev_ok = (t > 0)     && (mask[t - 1] == 0);
    bool next_ok = (t < T - 1) && (mask[t + 1] == 0);
    int prev = prev_ok ? z[t - 1] : 0;
    int next = next_ok ? z[t + 1] : 0;
    bool both = prev_ok && next_ok;

    const float* fr = logM  + (size_t)prev * KSYM;
    const float* br = logMT + (size_t)next * KSYM;
    // normalized-logits shift: log of WMMA-computed (M@M)[prev,next]
    float shift = both ? __logf(fmaxf(S[(size_t)prev * KSYM + next], 1e-30f))
                       : 0.0f;

    float best = -3.0e38f;
    int bestk = 0;
    for (int c0 = 0; c0 < KSYM; c0 += 32) {
        int k = c0 + lane;
        if (c0 + 64 < KSYM) {
            __builtin_prefetch(fr + c0 + 64, 0, 1);
            __builtin_prefetch(br + c0 + 64, 0, 1);
        }
        float s;
        if (prev_ok) {
            s = fr[k] - shift;
            if (both) s += br[k];
        } else {
            s = __logf(fmaxf(marg[k], 1e-30f));
        }
        uint32_t h = (uint32_t)t * 0x9E3779B9u + (uint32_t)k * 0x85EBCA6Bu;
        float g = -__logf(-__logf(pm_u01(h)));     // Gumbel(0,1)
        s += g;
        if (s > best || (s == best && k < bestk)) { best = s; bestk = k; }
    }
    // wave32 argmax reduction
#pragma unroll
    for (int off = 16; off > 0; off >>= 1) {
        float ob  = __shfl_xor(best,  off, 32);
        int   okk = __shfl_xor(bestk, off, 32);
        if (ob > best || (ob == best && okk < bestk)) { best = ob; bestk = okk; }
    }
    if (lane == 0) out[t] = bestk;
}

// ---------------------------------------------------------------------------
extern "C" void kernel_launch(void* const* d_in, const int* in_sizes, int n_in,
                              void* d_out, int out_size, void* d_ws, size_t ws_size,
                              hipStream_t stream) {
    const int*           z    = (const int*)d_in[0];
    const unsigned char* mask = (const unsigned char*)d_in[1];  // jnp.bool_
    const float*         M    = (const float*)d_in[2];
    const float*         marg = (const float*)d_in[3];
    int* out = (int*)d_out;
    const int T = in_sizes[0];

    // Workspace carve-up (~4.2 MB)
    char* ws = (char*)d_ws;
    __bf16* Mbf   = (__bf16*)ws;  ws += (size_t)KSYM * KSYM * sizeof(__bf16);
    __bf16* MTbf  = (__bf16*)ws;  ws += (size_t)KSYM * KSYM * sizeof(__bf16);
    float*  logM  = (float*)ws;   ws += (size_t)KSYM * KSYM * sizeof(float);
    float*  logMT = (float*)ws;   ws += (size_t)KSYM * KSYM * sizeof(float);
    float*  S     = (float*)ws;   ws += (size_t)KSYM * KSYM * sizeof(float);

    // 1) tables
    pm_prep<<<(KSYM * KSYM) / 256, 256, 0, stream>>>(M, Mbf, MTbf, logM, logMT);
    // 2) S = M @ M  (32 tile rows x 4 col-groups = 128 blocks of 8 waves)
    pm_gemm_bf16<<<128, 256, 0, stream>>>(Mbf, MTbf, S);
    // 3) one wave32 per position, 8 waves per 256-thread block
    pm_sample<<<(T + 7) / 8, 256, 0, stream>>>(z, mask, marg, logM, logMT, S, out, T);
}
